// esim_4037269258405
// MI455X (gfx1250) — matrix operand order repaired and checked by reading, not compiled, not run on hardware
//
#include <hip/hip_runtime.h>

// ============================================================================
// ESIM forward for MI455X (gfx1250, wave32).
// WMMA (v_wmma_f32_16x16x32_f16) used for: per-timestep cross-attention
// (A = X1.X2^T, alpha.X2, beta.X1) and the big match->dense GEMM.
// Serial LSTM recurrences run as one workgroup per (batch, direction) with
// Wh/h/c resident in LDS.
// ============================================================================

typedef __attribute__((ext_vector_type(16))) _Float16 v16h;
typedef __attribute__((ext_vector_type(8)))  float    v8f;

#define B_    128
#define S_    512
#define E_    50
#define U_    50
#define G_    200      // 4U gates
#define DF_   100      // 2U bilstm feature width
#define ROWS_ (B_ * S_)

__device__ __forceinline__ float sigmoidf_(float x) { return 1.f / (1.f + __expf(-x)); }

// ---- WMMA fragment helpers (wave32; contiguous k per lane half, same order for A and B) ----------
// A fragment 16x32 f16: lane L<16 -> row m0+L, K k0..k0+15 contiguous;
//                       lane L>=16 -> row m0+L-16, K k0+16..k0+31.
__device__ __forceinline__ v16h load_a16(const _Float16* base, int ld, int m0, int k0, int lane) {
  int m  = m0 + (lane & 15);
  int kb = k0 + ((lane >> 4) << 4);
  const _Float16* p = base + m * ld + kb;
  v16h a;
#pragma unroll
  for (int i = 0; i < 16; ++i) a[i] = p[i];
  return a;
}
// B fragment 32x16 f16 from an N-major buffer (layout [n][k]):
// lane L<16 -> col n0+L, K k0..k0+15 contiguous; lane>=16 -> K k0+16..k0+31.
__device__ __forceinline__ v16h load_bT16(const _Float16* baseT, int ld, int n0, int k0, int lane) {
  int n  = n0 + (lane & 15);
  int kb = k0 + ((lane >> 4) << 4);
  const _Float16* p = baseT + n * ld + kb;
  v16h b;
#pragma unroll
  for (int i = 0; i < 16; ++i) b[i] = p[i];
  return b;
}
__device__ __forceinline__ v8f wmma16(v16h a, v16h b, v8f c) {
  return __builtin_amdgcn_wmma_f32_16x16x32_f16(false, a, false, b, (short)0, c, false, false);
}

// ============================================================================
// xz = input @ Wx + b      (fused embedding+mask when toks != nullptr)
// ============================================================================
__global__ void xz_kernel(const int* __restrict__ toks, const float* __restrict__ emb,
                          const float* __restrict__ densein, int Din,
                          const float* __restrict__ Wx, const float* __restrict__ bias,
                          float* __restrict__ xz) {
  int idx = blockIdx.x * blockDim.x + threadIdx.x;
  if (idx >= ROWS_ * G_) return;
  int row = idx / G_, k = idx % G_;
  float acc = bias[k];
  if (toks) {
    int t = toks[row];
    if (t != 0) {
      const float* e = emb + (size_t)t * E_;
#pragma unroll 10
      for (int d = 0; d < E_; ++d) acc += e[d] * Wx[d * G_ + k];
    }
  } else {
    const float* e = densein + (size_t)row * Din;
#pragma unroll 10
    for (int d = 0; d < Din; ++d) acc += e[d] * Wx[d * G_ + k];
  }
  xz[(size_t)row * G_ + k] = acc;
}

// ============================================================================
// LSTM recurrence: one block per (batch, direction). Wh/h/c/z in LDS.
// Keras gate order i,f,g,o.  Output masked by token!=0 when toks != nullptr.
// ============================================================================
__global__ void __launch_bounds__(256)
lstm_rec_kernel(const float* __restrict__ xzf, const float* __restrict__ xzb,
                const float* __restrict__ Whf, const float* __restrict__ Whb,
                const int* __restrict__ toks, float* __restrict__ out) {
  __shared__ float Wh[U_ * G_];
  __shared__ float hL[U_], cL[U_], zL[G_];
  const int b = blockIdx.x, dir = blockIdx.y, tid = threadIdx.x;
  const float* xz  = dir ? xzb : xzf;
  const float* Whg = dir ? Whb : Whf;
  for (int i = tid; i < U_ * G_; i += 256) Wh[i] = Whg[i];
  if (tid < U_) { hL[tid] = 0.f; cL[tid] = 0.f; }
  __syncthreads();
  for (int t = 0; t < S_; ++t) {
    int src = dir ? (S_ - 1 - t) : t;
    if (tid < G_) {
      float acc = xz[((size_t)b * S_ + src) * G_ + tid];
#pragma unroll
      for (int j = 0; j < U_; ++j) acc += hL[j] * Wh[j * G_ + tid];
      zL[tid] = acc;
    }
    __syncthreads();
    if (tid < U_) {
      float iv = sigmoidf_(zL[tid]);
      float fv = sigmoidf_(zL[U_ + tid]);
      float gv = tanhf(zL[2 * U_ + tid]);
      float ov = sigmoidf_(zL[3 * U_ + tid]);
      float cn = fv * cL[tid] + iv * gv;
      cL[tid] = cn;
      float hn = ov * tanhf(cn);
      hL[tid] = hn;
      float mval = toks ? ((toks[(size_t)b * S_ + t] != 0) ? 1.f : 0.f) : 1.f;
      out[((size_t)b * S_ + t) * DF_ + dir * U_ + tid] = hn * mval;
    }
    __syncthreads();
  }
}

// ============================================================================
// Cross-attention per timestep s: A = X1.X2^T (128x128), masked softmaxes,
// d1 = alpha.X2, d2 = beta.X1.  All GEMMs on WMMA f16 with f32 accumulate.
// One workgroup (8 waves) per timestep; ~225 KB dynamic LDS (1 WG / WGP).
// ============================================================================
__global__ void __launch_bounds__(256)
attn_kernel(const float* __restrict__ enc1, const float* __restrict__ enc2,
            const int* __restrict__ x1, const int* __restrict__ x2,
            float* __restrict__ d1, float* __restrict__ d2) {
  extern __shared__ char smem[];
  _Float16* X1h = (_Float16*)smem;           // [128][128] row=i, col=k
  _Float16* X2h = X1h + 128 * 128;           // [128][128] row=j, col=k
  _Float16* X1t = X2h + 128 * 128;           // [128][128] row=k, col=i
  _Float16* X2t = X1t + 128 * 128;           // [128][128] row=k, col=j
  float*    A   = (float*)(X2t + 128 * 128); // [128][128]
  _Float16* P   = (_Float16*)(A + 128 * 128);// probs (alpha then beta)
  float*    M1  = (float*)(P + 128 * 128);   // [128]
  float*    M2  = M1 + 128;

  const int s = blockIdx.x;
  const int tid = threadIdx.x, lane = tid & 31, w = tid >> 5;

  for (int idx = tid; idx < 128 * 128; idx += 256) {
    int i = idx >> 7, k = idx & 127;
    float v1 = (k < DF_) ? enc1[((size_t)i * S_ + s) * DF_ + k] : 0.f;
    float v2 = (k < DF_) ? enc2[((size_t)i * S_ + s) * DF_ + k] : 0.f;
    _Float16 h1 = (_Float16)v1, h2 = (_Float16)v2;
    X1h[idx] = h1; X2h[idx] = h2;
    X1t[k * 128 + i] = h1; X2t[k * 128 + i] = h2;
  }
  if (tid < 128)       M1[tid]       = (x1[(size_t)tid * S_ + s] != 0) ? 1.f : 0.f;
  else                 M2[tid - 128] = (x2[(size_t)(tid - 128) * S_ + s] != 0) ? 1.f : 0.f;
  __syncthreads();

  // ---- A = X1 @ X2^T : wave w owns row-tile w, all 8 col-tiles -------------
  {
    v16h a0 = load_a16(X1h, 128, w * 16, 0,  lane);
    v16h a1 = load_a16(X1h, 128, w * 16, 32, lane);
    v16h a2 = load_a16(X1h, 128, w * 16, 64, lane);
    v16h a3 = load_a16(X1h, 128, w * 16, 96, lane);
    for (int nt = 0; nt < 8; ++nt) {
      v8f acc = {};
      acc = wmma16(a0, load_bT16(X2h, 128, nt * 16, 0,  lane), acc);
      acc = wmma16(a1, load_bT16(X2h, 128, nt * 16, 32, lane), acc);
      acc = wmma16(a2, load_bT16(X2h, 128, nt * 16, 64, lane), acc);
      acc = wmma16(a3, load_bT16(X2h, 128, nt * 16, 96, lane), acc);
      int col = nt * 16 + (lane & 15);
      int rb  = w * 16 + ((lane >> 4) << 3);
#pragma unroll
      for (int r = 0; r < 8; ++r) A[(rb + r) * 128 + col] = acc[r];
    }
  }
  __syncthreads();

  // ---- alpha (softmax over j, mask m2) ------------------------------------
  if (tid < 128) {
    int i = tid;
    float mx = -3.402823e38f;
    for (int j = 0; j < 128; ++j) mx = fmaxf(mx, A[i * 128 + j]);
    float sum = 0.f;
    for (int j = 0; j < 128; ++j) {
      float wv = __expf(A[i * 128 + j] - mx) * M2[j];
      sum += wv;
      P[i * 128 + j] = (_Float16)wv;
    }
    float inv = 1.f / (sum + 1e-8f);
    for (int j = 0; j < 128; ++j) P[i * 128 + j] = (_Float16)((float)P[i * 128 + j] * inv);
  }
  __syncthreads();

  // ---- d1 = alpha @ X2  (B operand from X2t, N-major over d) --------------
  {
    v16h a0 = load_a16(P, 128, w * 16, 0,  lane);
    v16h a1 = load_a16(P, 128, w * 16, 32, lane);
    v16h a2 = load_a16(P, 128, w * 16, 64, lane);
    v16h a3 = load_a16(P, 128, w * 16, 96, lane);
    for (int nt = 0; nt < 7; ++nt) {
      v8f acc = {};
      acc = wmma16(a0, load_bT16(X2t, 128, nt * 16, 0,  lane), acc);
      acc = wmma16(a1, load_bT16(X2t, 128, nt * 16, 32, lane), acc);
      acc = wmma16(a2, load_bT16(X2t, 128, nt * 16, 64, lane), acc);
      acc = wmma16(a3, load_bT16(X2t, 128, nt * 16, 96, lane), acc);
      int dcol = nt * 16 + (lane & 15);
      if (dcol < DF_) {
        int rb = w * 16 + ((lane >> 4) << 3);
#pragma unroll
        for (int r = 0; r < 8; ++r)
          d1[((size_t)(rb + r) * S_ + s) * DF_ + dcol] = acc[r];
      }
    }
  }
  __syncthreads();

  // ---- beta (softmax over i of A^T, mask m1) ------------------------------
  if (tid < 128) {
    int j = tid;
    float mx = -3.402823e38f;
    for (int i = 0; i < 128; ++i) mx = fmaxf(mx, A[i * 128 + j]);
    float sum = 0.f;
    for (int i = 0; i < 128; ++i) {
      float wv = __expf(A[i * 128 + j] - mx) * M1[i];
      sum += wv;
      P[j * 128 + i] = (_Float16)wv;
    }
    float inv = 1.f / (sum + 1e-8f);
    for (int i = 0; i < 128; ++i) P[j * 128 + i] = (_Float16)((float)P[j * 128 + i] * inv);
  }
  __syncthreads();

  // ---- d2 = beta @ X1  (B operand from X1t) -------------------------------
  {
    v16h a0 = load_a16(P, 128, w * 16, 0,  lane);
    v16h a1 = load_a16(P, 128, w * 16, 32, lane);
    v16h a2 = load_a16(P, 128, w * 16, 64, lane);
    v16h a3 = load_a16(P, 128, w * 16, 96, lane);
    for (int nt = 0; nt < 7; ++nt) {
      v8f acc = {};
      acc = wmma16(a0, load_bT16(X1t, 128, nt * 16, 0,  lane), acc);
      acc = wmma16(a1, load_bT16(X1t, 128, nt * 16, 32, lane), acc);
      acc = wmma16(a2, load_bT16(X1t, 128, nt * 16, 64, lane), acc);
      acc = wmma16(a3, load_bT16(X1t, 128, nt * 16, 96, lane), acc);
      int dcol = nt * 16 + (lane & 15);
      if (dcol < DF_) {
        int rb = w * 16 + ((lane >> 4) << 3);
#pragma unroll
        for (int r = 0; r < 8; ++r)
          d2[((size_t)(rb + r) * S_ + s) * DF_ + dcol] = acc[r];
      }
    }
  }
}

// ============================================================================
// map = relu([enc, d, enc*d, enc-d] @ dense_W + dense_b) via WMMA.
// Block = 128 threads (4 waves), 64 rows per block.  K padded 400->416,
// N padded 100->112.  ~146 KB dynamic LDS.
// ============================================================================
#define KD_ 416
#define ND_ 112
__global__ void __launch_bounds__(128)
matchdense_kernel(const float* __restrict__ enc, const float* __restrict__ dmat,
                  const float* __restrict__ W, const float* __restrict__ bias,
                  float* __restrict__ mapout) {
  extern __shared__ char smem[];
  _Float16* Mh = (_Float16*)smem;      // [64][416] match rows, f16
  _Float16* Wn = Mh + 64 * KD_;        // [112][416] weights, N-major f16
  const int tid = threadIdx.x, lane = tid & 31, w = tid >> 5;
  const size_t r0 = (size_t)blockIdx.x * 64;

  for (int idx = tid; idx < 64 * KD_; idx += 128) {
    int r = idx / KD_, c = idx % KD_;
    float v = 0.f;
    if (c < 400) {
      int grp = c / 100, cc = c % 100;
      size_t g = (r0 + r) * DF_ + cc;
      float e = enc[g], dd = dmat[g];
      v = (grp == 0) ? e : (grp == 1) ? dd : (grp == 2) ? e * dd : e - dd;
    }
    Mh[idx] = (_Float16)v;
  }
  for (int idx = tid; idx < ND_ * KD_; idx += 128) {
    int n = idx / KD_, k = idx % KD_;
    Wn[idx] = (_Float16)((n < 100 && k < 400) ? W[k * 100 + n] : 0.f);
  }
  __syncthreads();

  v16h afr[13];
#pragma unroll
  for (int kt = 0; kt < 13; ++kt) afr[kt] = load_a16(Mh, KD_, w * 16, kt * 32, lane);
  for (int nt = 0; nt < 7; ++nt) {
    v8f acc = {};
#pragma unroll
    for (int kt = 0; kt < 13; ++kt)
      acc = wmma16(afr[kt], load_bT16(Wn, KD_, nt * 16, kt * 32, lane), acc);
    int n = nt * 16 + (lane & 15);
    if (n < 100) {
      float bv = bias[n];
      int rb = w * 16 + ((lane >> 4) << 3);
#pragma unroll
      for (int r = 0; r < 8; ++r)
        mapout[(r0 + rb + r) * DF_ + n] = fmaxf(acc[r] + bv, 0.f);
    }
  }
}

// ============================================================================
// Masked mean / max pooling -> logit[b, 400] = [p1_mean, p1_max, p2_mean, p2_max]
// ============================================================================
__global__ void pool_kernel(const float* __restrict__ cmp1, const float* __restrict__ cmp2,
                            const int* __restrict__ x1, const int* __restrict__ x2,
                            float* __restrict__ logit) {
  int idx = blockIdx.x * blockDim.x + threadIdx.x;
  if (idx >= 2 * B_ * DF_) return;
  int sel = idx / (B_ * DF_);
  int rem = idx % (B_ * DF_);
  int b = rem / DF_, k = rem % DF_;
  const float* cmp = sel ? cmp2 : cmp1;
  const int*   xt  = sel ? x2 : x1;
  float sum = 0.f, mx = -3.402823e38f, msum = 0.f;
  for (int t = 0; t < S_; ++t) {
    float mval = (xt[(size_t)b * S_ + t] != 0) ? 1.f : 0.f;
    float v = cmp[((size_t)b * S_ + t) * DF_ + k] * mval;
    sum += v; mx = fmaxf(mx, v); msum += mval;
  }
  logit[b * 400 + sel * 200 + k]       = sum / msum;
  logit[b * 400 + sel * 200 + 100 + k] = mx;
}

// ============================================================================
// Head: tanh(logit @ dense1_W + b) @ out_W + out_b -> softmax(2)
// ============================================================================
__global__ void __launch_bounds__(128)
head_kernel(const float* __restrict__ logit, const float* __restrict__ W1,
            const float* __restrict__ b1, const float* __restrict__ W2,
            const float* __restrict__ b2, float* __restrict__ out) {
  __shared__ float h1[100];
  __shared__ float l2[2];
  const int b = blockIdx.x, tid = threadIdx.x;
  if (tid < 100) {
    float acc = b1[tid];
    for (int kk = 0; kk < 400; ++kk) acc += logit[b * 400 + kk] * W1[kk * 100 + tid];
    h1[tid] = tanhf(acc);
  }
  __syncthreads();
  if (tid < 2) {
    float acc = b2[tid];
    for (int j = 0; j < 100; ++j) acc += h1[j] * W2[j * 2 + tid];
    l2[tid] = acc;
  }
  __syncthreads();
  if (tid < 2) {
    float mx = fmaxf(l2[0], l2[1]);
    float e0 = __expf(l2[0] - mx), e1 = __expf(l2[1] - mx);
    out[b * 2 + tid] = (tid ? e1 : e0) / (e0 + e1);
  }
}

// ============================================================================
extern "C" void kernel_launch(void* const* d_in, const int* in_sizes, int n_in,
                              void* d_out, int out_size, void* d_ws, size_t ws_size,
                              hipStream_t stream) {
  const int*   x1       = (const int*)d_in[0];
  const int*   x2       = (const int*)d_in[1];
  const float* emb      = (const float*)d_in[2];
  const float* b1f_Wx   = (const float*)d_in[3];
  const float* b1f_Wh   = (const float*)d_in[4];
  const float* b1f_b    = (const float*)d_in[5];
  const float* b1b_Wx   = (const float*)d_in[6];
  const float* b1b_Wh   = (const float*)d_in[7];
  const float* b1b_b    = (const float*)d_in[8];
  const float* dense_W  = (const float*)d_in[9];
  const float* dense_b  = (const float*)d_in[10];
  const float* b2f_Wx   = (const float*)d_in[11];
  const float* b2f_Wh   = (const float*)d_in[12];
  const float* b2f_b    = (const float*)d_in[13];
  const float* b2b_Wx   = (const float*)d_in[14];
  const float* b2b_Wh   = (const float*)d_in[15];
  const float* b2b_b    = (const float*)d_in[16];
  const float* dense1_W = (const float*)d_in[17];
  const float* dense1_b = (const float*)d_in[18];
  const float* out_W    = (const float*)d_in[19];
  const float* out_b    = (const float*)d_in[20];
  float* outp = (float*)d_out;

  float* ws = (float*)d_ws;
  size_t o = 0;
  float* xzf  = ws + o; o += (size_t)ROWS_ * G_;    // 13,107,200
  float* xzb  = ws + o; o += (size_t)ROWS_ * G_;
  float* enc1 = ws + o; o += (size_t)ROWS_ * DF_;   // 6,553,600
  float* enc2 = ws + o; o += (size_t)ROWS_ * DF_;
  float* d1   = ws + o; o += (size_t)ROWS_ * DF_;
  float* d2   = ws + o; o += (size_t)ROWS_ * DF_;
  float* mapb = ws + o; o += (size_t)ROWS_ * DF_;
  float* cmp1 = ws + o; o += (size_t)ROWS_ * DF_;
  float* cmp2 = ws + o; o += (size_t)ROWS_ * DF_;
  float* logit= ws + o; o += (size_t)B_ * 400;

  const int xzTotal = ROWS_ * G_;
  dim3 xzGrid((xzTotal + 255) / 256);
  const size_t attnLds = (size_t)4 * 128 * 128 * 2 + 128 * 128 * 4 + 128 * 128 * 2 + 2 * 128 * 4;
  const size_t mdLds   = (size_t)64 * KD_ * 2 + (size_t)ND_ * KD_ * 2;

  // ---- BiLSTM layer 1 on x1 ----
  xz_kernel<<<xzGrid, 256, 0, stream>>>(x1, emb, nullptr, E_, b1f_Wx, b1f_b, xzf);
  xz_kernel<<<xzGrid, 256, 0, stream>>>(x1, emb, nullptr, E_, b1b_Wx, b1b_b, xzb);
  lstm_rec_kernel<<<dim3(B_, 2), 256, 0, stream>>>(xzf, xzb, b1f_Wh, b1b_Wh, x1, enc1);
  // ---- BiLSTM layer 1 on x2 (shared weights) ----
  xz_kernel<<<xzGrid, 256, 0, stream>>>(x2, emb, nullptr, E_, b1f_Wx, b1f_b, xzf);
  xz_kernel<<<xzGrid, 256, 0, stream>>>(x2, emb, nullptr, E_, b1b_Wx, b1b_b, xzb);
  lstm_rec_kernel<<<dim3(B_, 2), 256, 0, stream>>>(xzf, xzb, b1f_Wh, b1b_Wh, x2, enc2);

  // ---- cross attention (WMMA) ----
  attn_kernel<<<S_, 256, attnLds, stream>>>(enc1, enc2, x1, x2, d1, d2);

  // ---- side 1: match+dense (WMMA) then BiLSTM layer 2 ----
  matchdense_kernel<<<ROWS_ / 64, 128, mdLds, stream>>>(enc1, d1, dense_W, dense_b, mapb);
  xz_kernel<<<xzGrid, 256, 0, stream>>>(nullptr, emb, mapb, DF_, b2f_Wx, b2f_b, xzf);
  xz_kernel<<<xzGrid, 256, 0, stream>>>(nullptr, emb, mapb, DF_, b2b_Wx, b2b_b, xzb);
  lstm_rec_kernel<<<dim3(B_, 2), 256, 0, stream>>>(xzf, xzb, b2f_Wh, b2b_Wh, nullptr, cmp1);

  // ---- side 2 ----
  matchdense_kernel<<<ROWS_ / 64, 128, mdLds, stream>>>(enc2, d2, dense_W, dense_b, mapb);
  xz_kernel<<<xzGrid, 256, 0, stream>>>(nullptr, emb, mapb, DF_, b2f_Wx, b2f_b, xzf);
  xz_kernel<<<xzGrid, 256, 0, stream>>>(nullptr, emb, mapb, DF_, b2b_Wx, b2b_b, xzb);
  lstm_rec_kernel<<<dim3(B_, 2), 256, 0, stream>>>(xzf, xzb, b2f_Wh, b2b_Wh, nullptr, cmp2);

  // ---- pooling + head ----
  pool_kernel<<<(2 * B_ * DF_ + 255) / 256, 256, 0, stream>>>(cmp1, cmp2, x1, x2, logit);
  head_kernel<<<B_, 128, 0, stream>>>(logit, dense1_W, dense1_b, out_W, out_b, outp);
}
